// MultiHeadAttention_91190745628911
// MI455X (gfx1250) — matrix operand order repaired and checked by reading, not compiled, not run on hardware
//
#include <hip/hip_runtime.h>
#include <hip/hip_bf16.h>

// ---------------------------------------------------------------------------
// Fused MHA forward for gfx1250 (MI455X): LN -> QKV gemm -> flash attention
// -> out-proj. Matmuls via v_wmma_f32_16x16x32_bf16 (wave32); dense GEMMs and
// attention K/V staging use a double-buffered Tensor Data Mover pipeline
// (tensor_load_to_lds + s_wait_tensorcnt), fragments come from LDS.
// ---------------------------------------------------------------------------

#define SEQ     2048
#define BATCH   2
#define HIDDEN  1024
#define NHEAD   16
#define HDIM    64
#define NTOK    (SEQ * BATCH)        // 4096 tokens (t = s*BATCH + b)
#define QKVN    (3 * HIDDEN)         // 3072

typedef __attribute__((ext_vector_type(16))) __bf16        v16bf;
typedef __attribute__((ext_vector_type(8)))  float         v8f;
typedef __attribute__((ext_vector_type(4)))  unsigned int  v4u;
typedef __attribute__((ext_vector_type(8)))  int           v8i_t;
typedef __attribute__((ext_vector_type(4)))  int           v4i_t;

static __device__ __forceinline__ v8f zero8() {
  v8f z = {0.f, 0.f, 0.f, 0.f, 0.f, 0.f, 0.f, 0.f};
  return z;
}

static __device__ __forceinline__ v8f bwmma(v16bf a, v16bf b, v8f c) {
  // D = A(16x32 bf16) * B(32x16 bf16) + C(16x16 f32)
  return __builtin_amdgcn_wmma_f32_16x16x32_bf16(
      false, a, false, b, (short)0, c, false, false);
}

// A-matrix fragment (16x32 bf16): lane = row M (lane&15); contiguous
// K run = half*16 + 0..15 with half = lane>>4.
static __device__ __forceinline__ v16bf load_afrag(const __bf16* rowp, int k0) {
  union { v16bf v; uint4 u[2]; } f;
  f.u[0] = *(const uint4*)(rowp + k0);
  f.u[1] = *(const uint4*)(rowp + k0 + 8);
  return f.v;
}

// B-matrix fragment (32x16 bf16): lane = column N (lane&15); contiguous
// K run = half*16 + 0..15. p points at (column base + kstep + half*16).
static __device__ __forceinline__ v16bf load_bfrag(const __bf16* p) {
  union { v16bf v; uint4 u[2]; } f;
  f.u[0] = *(const uint4*)(p);
  f.u[1] = *(const uint4*)(p + 8);
  return f.v;
}

static __device__ __forceinline__ float rowred_max(float v) {
#pragma unroll
  for (int m = 1; m <= 8; m <<= 1) v = fmaxf(v, __shfl_xor(v, m, 32));
  return v;
}
static __device__ __forceinline__ float rowred_sum(float v) {
#pragma unroll
  for (int m = 1; m <= 8; m <<= 1) v += __shfl_xor(v, m, 32);
  return v;
}

// ---------------------------------------------------------------------------
// Tensor Data Mover: 2-D tile load Global -> LDS, bf16 elements.
// D# packing per CDNA5 ISA 8.3/8.4 (count=1, type=2, data_size=2B).
// ---------------------------------------------------------------------------
#if __has_builtin(__builtin_amdgcn_tensor_load_to_lds)
#define USE_TDM 1
static __device__ __forceinline__ void tdm_load_2d(unsigned lds_off,
                                                   const void* gaddr,
                                                   unsigned tile_d0,
                                                   unsigned tile_d1,
                                                   unsigned row_stride_elems) {
  const unsigned long long ga = (unsigned long long)(uintptr_t)gaddr;
  v4u g0;
  g0[0] = 1u;                                           // count=1 (valid D#)
  g0[1] = lds_off;                                      // lds_addr [63:32]
  g0[2] = (unsigned)ga;                                 // global_addr lo
  g0[3] = ((unsigned)(ga >> 32) & 0x01FFFFFFu)          // global_addr hi
          | (2u << 30);                                 // type=2 ("image")
  v8i_t g1;
  g1[0] = (int)(1u << 16);                              // data_size=1 -> 2B
  g1[1] = (int)((tile_d0 & 0xFFFFu) << 16);             // tensor_dim0[15:0]
  g1[2] = (int)(((tile_d0 >> 16) & 0xFFFFu)             // tensor_dim0[31:16]
                | ((tile_d1 & 0xFFFFu) << 16));         // tensor_dim1[15:0]
  g1[3] = (int)(((tile_d1 >> 16) & 0xFFFFu)             // tensor_dim1[31:16]
                | ((tile_d0 & 0xFFFFu) << 16));         // tile_dim0
  g1[4] = (int)(tile_d1 & 0xFFFFu);                     // tile_dim1 (dim2=0)
  g1[5] = (int)row_stride_elems;                        // dim0_stride lo
  g1[6] = 0;                                            // dim0_stride hi
  g1[7] = 0;                                            // dim1_stride (unused)
  v4i_t z4 = {0, 0, 0, 0};
#if defined(__clang_major__) && (__clang_major__ >= 23)
  v8i_t z8 = {0, 0, 0, 0, 0, 0, 0, 0};
  __builtin_amdgcn_tensor_load_to_lds(g0, g1, z4, z4, z8, 0);
#else
  __builtin_amdgcn_tensor_load_to_lds(g0, g1, z4, z4, 0);
#endif
}
#else
#define USE_TDM 0
#endif

static __device__ __forceinline__ unsigned lds_byte_off(const void* p) {
  return (unsigned)(uintptr_t)p;   // low 32 bits of generic addr = LDS offset
}

// ---------------------------------------------------------------------------
// Shared GEMM core: C[tile 32M x 512N] += A[32 x 1024] * Bt^T.
// 8 waves; wave = 16M x 128N (8 accumulators / A-fragment).
// TDM double-buffered LDS pipeline (wave 0 drives the TDM).
// All 8 B fragments are loaded before the WMMA burst (one dscnt wait).
// ---------------------------------------------------------------------------
static __device__ __forceinline__ void gemm_core(const __bf16* __restrict__ A,
                                                 const __bf16* __restrict__ Bt,
                                                 int mBlock, int nBlock,
                                                 v8f acc[8]) {
  const int lane = threadIdx.x & 31, w = threadIdx.x >> 5;
  const int half = lane >> 4, l15 = lane & 15;
  const int waveM = w >> 2, waveN = w & 3;

#if USE_TDM
  __shared__ __align__(16) __bf16 Ab[2][32 * 32];     //  2 x 2 KB
  __shared__ __align__(16) __bf16 Bb[2][512 * 32];    //  2 x 32 KB
  const __bf16* aTile = A + (size_t)mBlock * HIDDEN;
  const __bf16* bTile = Bt + (size_t)nBlock * HIDDEN;

  if (w == 0) {
    tdm_load_2d(lds_byte_off(&Ab[0][0]), aTile, 32, 32, HIDDEN);
    tdm_load_2d(lds_byte_off(&Bb[0][0]), bTile, 32, 512, HIDDEN);
  }
  for (int step = 0; step < HIDDEN / 32; ++step) {
    const int cur = step & 1;
    if (w == 0) __builtin_amdgcn_s_wait_tensorcnt(0);
    __syncthreads();                       // buf[cur] visible to all waves
    if (w == 0 && step + 1 < HIDDEN / 32) {
      const int k = (step + 1) * 32;
      tdm_load_2d(lds_byte_off(&Ab[cur ^ 1][0]), aTile + k, 32, 32, HIDDEN);
      tdm_load_2d(lds_byte_off(&Bb[cur ^ 1][0]), bTile + k, 32, 512, HIDDEN);
    }
    const v16bf a = load_afrag(&Ab[cur][(waveM * 16 + l15) * 32], half * 16);
    const __bf16* bt = &Bb[cur][(size_t)waveN * 128 * 32];
    v16bf b[8];
#pragma unroll
    for (int nt = 0; nt < 8; ++nt)
      b[nt] = load_bfrag(bt + (nt * 16 + l15) * 32 + half * 16);
#pragma unroll
    for (int nt = 0; nt < 8; ++nt)
      acc[nt] = bwmma(a, b[nt], acc[nt]);
    __syncthreads();                       // done reading buf[cur]
  }
#else
  // Fallback: direct global-memory fragment streaming.
  const __bf16* arow = A + (size_t)(mBlock + waveM * 16 + l15) * HIDDEN;
  const __bf16* bbase = Bt + (size_t)(nBlock + waveN * 128 + l15) * HIDDEN;
  for (int k = 0; k < HIDDEN; k += 32) {
    const v16bf a = load_afrag(arow, k + half * 16);
    v16bf b[8];
#pragma unroll
    for (int nt = 0; nt < 8; ++nt)
      b[nt] = load_bfrag(bbase + (size_t)nt * 16 * HIDDEN + k + half * 16);
#pragma unroll
    for (int nt = 0; nt < 8; ++nt)
      acc[nt] = bwmma(a, b[nt], acc[nt]);
  }
#endif
}

// ---------------------------------------------------------------------------
// Kernel 1: LayerNorm over HIDDEN=1024, output bf16. One block per token.
// ---------------------------------------------------------------------------
__global__ __launch_bounds__(256)
void ln_bf16_kernel(const float* __restrict__ x, const float* __restrict__ gamma,
                    const float* __restrict__ beta, __bf16* __restrict__ y) {
  __shared__ float s_sum[8], s_sq[8];
  const int t   = blockIdx.x;
  const int tid = threadIdx.x;
  const float4 v = ((const float4*)(x + (size_t)t * HIDDEN))[tid];
  float s = v.x + v.y + v.z + v.w;
  float q = v.x * v.x + v.y * v.y + v.z * v.z + v.w * v.w;
#pragma unroll
  for (int m = 1; m < 32; m <<= 1) {
    s += __shfl_xor(s, m, 32);
    q += __shfl_xor(q, m, 32);
  }
  const int w = tid >> 5;
  if ((tid & 31) == 0) { s_sum[w] = s; s_sq[w] = q; }
  __syncthreads();
  float S = 0.f, Q = 0.f;
#pragma unroll
  for (int i = 0; i < 8; ++i) { S += s_sum[i]; Q += s_sq[i]; }
  const float mean = S * (1.f / HIDDEN);
  const float var  = Q * (1.f / HIDDEN) - mean * mean;
  const float inv  = rsqrtf(var + 1e-6f);
  const float4 g = ((const float4*)gamma)[tid];
  const float4 b = ((const float4*)beta)[tid];
  __bf16* yp = y + (size_t)t * HIDDEN + tid * 4;
  yp[0] = (__bf16)((v.x - mean) * inv * g.x + b.x);
  yp[1] = (__bf16)((v.y - mean) * inv * g.y + b.y);
  yp[2] = (__bf16)((v.z - mean) * inv * g.z + b.z);
  yp[3] = (__bf16)((v.w - mean) * inv * g.w + b.w);
}

// ---------------------------------------------------------------------------
// Weight transposes (fp32 -> bf16) so B fragments are contiguous K runs.
// ---------------------------------------------------------------------------
__global__ __launch_bounds__(256)
void wqkv_T_kernel(const float* __restrict__ w, __bf16* __restrict__ wt) {
  const int idx = blockIdx.x * 256 + threadIdx.x;       // n*1024 + e
  if (idx >= QKVN * HIDDEN) return;
  const int n = idx >> 10, e = idx & 1023;
  wt[idx] = (__bf16)w[(size_t)e * QKVN + n];
}

__global__ __launch_bounds__(256)
void wout_T_kernel(const float* __restrict__ w, __bf16* __restrict__ wt) {
  const int idx = blockIdx.x * 256 + threadIdx.x;       // e*1024 + f
  if (idx >= HIDDEN * HIDDEN) return;
  const int e = idx >> 10, f = idx & 1023;
  wt[idx] = (__bf16)w[(size_t)f * HIDDEN + e];
}

// ---------------------------------------------------------------------------
// Kernel 3: QKV GEMM.  C[4096 x 3072] = LN * WqkvT^T; scatter into
// Q[b,h,s,d], K[b,h,s,d], Vt[b,h,d,s] as bf16.
// ---------------------------------------------------------------------------
__global__ __launch_bounds__(256)
void qkv_gemm_kernel(const __bf16* __restrict__ A, const __bf16* __restrict__ Bt,
                     __bf16* __restrict__ qb, __bf16* __restrict__ kb,
                     __bf16* __restrict__ vtb) {
  const int lane = threadIdx.x & 31, w = threadIdx.x >> 5;
  const int half = lane >> 4, l15 = lane & 15;
  const int mBlock = blockIdx.y * 32;
  const int nBlock = blockIdx.x * 512;

  v8f acc[8] = {zero8(), zero8(), zero8(), zero8(),
                zero8(), zero8(), zero8(), zero8()};
  gemm_core(A, Bt, mBlock, nBlock, acc);

  const int mBase = mBlock + (w >> 2) * 16;
  const int nBase = nBlock + (w & 3) * 128;
#pragma unroll
  for (int nt = 0; nt < 8; ++nt) {
    const int n   = nBase + nt * 16 + l15;
    const int sel = n >> 10;
    const int f   = n & 1023;
    const int h   = f >> 6;
    const int d   = f & 63;
#pragma unroll
    for (int r = 0; r < 8; ++r) {
      const int m  = mBase + half * 8 + r;
      const int s  = m >> 1;          // token t = s*BATCH + b
      const int bb = m & 1;
      const size_t bh = (size_t)(bb * NHEAD + h);
      const __bf16 val = (__bf16)acc[nt][r];
      if (sel == 0)      qb[(bh * SEQ + s) * HDIM + d]  = val;
      else if (sel == 1) kb[(bh * SEQ + s) * HDIM + d]  = val;
      else               vtb[(bh * HDIM + d) * SEQ + s] = val;
    }
  }
}

// ---------------------------------------------------------------------------
// Kernel 4: flash attention. All 8 waves of a block share one (b,h); wave w
// owns q-tile (blockIdx.x & 15)*8 + w. Per 32-key step the K tile (32x64,
// contiguous) and V tile (64 d-rows x 32 s, stride SEQ) are staged into LDS
// by a TDM double-buffer driven by wave 0 and consumed via ds_load_b128.
// Logits: 4 WMMA; mask; online softmax; P repacked C-layout -> A-layout via
// wave-private LDS; PV: 4 WMMA into a 16x64 f32 accumulator.
// ---------------------------------------------------------------------------
__global__ __launch_bounds__(256)
void attn_kernel(const __bf16* __restrict__ qb, const __bf16* __restrict__ kb,
                 const __bf16* __restrict__ vtb, const unsigned char* __restrict__ mask,
                 __bf16* __restrict__ ctxb) {
  __shared__ __align__(16) __bf16 pshare[8][16 * 32];   // per-wave P tile, 8 KB
#if USE_TDM
  __shared__ __align__(16) __bf16 Kb[2][32 * HDIM];     // 2 x 4 KB
  __shared__ __align__(16) __bf16 Vb[2][HDIM * 32];     // 2 x 4 KB
#endif
  const int lane = threadIdx.x & 31, w = threadIdx.x >> 5;
  const int half = lane >> 4, l15 = lane & 15;
  const int bh = blockIdx.x >> 4;               // 16 blocks per (b,h)
  const int qt = (blockIdx.x & 15) * 8 + w;
  const int b  = bh >> 4;
  const int h  = bh & 15;
  const int qBase = qt * 16;

  const __bf16* qp = qb + ((size_t)bh * SEQ + qBase + l15) * HDIM;
  const v16bf aq0 = load_afrag(qp, half * 16);        // d 0..31
  const v16bf aq1 = load_afrag(qp, 32 + half * 16);   // d 32..63
  const unsigned char* mrow = mask + (size_t)b * SEQ * SEQ;
  __bf16* pw = &pshare[w][0];

  v8f o[4] = {zero8(), zero8(), zero8(), zero8()};
  float mi[8], li[8];
#pragma unroll
  for (int r = 0; r < 8; ++r) { mi[r] = -3e38f; li[r] = 0.f; }

#if USE_TDM
  const __bf16* kTile = kb + (size_t)bh * SEQ * HDIM;   // [s][d], d contiguous
  const __bf16* vTile = vtb + (size_t)bh * HDIM * SEQ;  // [d][s], s contiguous
  if (w == 0) {
    tdm_load_2d(lds_byte_off(&Kb[0][0]), kTile, 32 * HDIM, 1, 32 * HDIM);
    tdm_load_2d(lds_byte_off(&Vb[0][0]), vTile, 32, HDIM, SEQ);
  }
#endif

  for (int j = 0; j < SEQ; j += 32) {
    // ---- stage K/V tiles for this step (TDM double buffer)
#if USE_TDM
    const int cur = (j >> 5) & 1;
    if (w == 0) __builtin_amdgcn_s_wait_tensorcnt(0);
    __syncthreads();                       // buf[cur] visible to all waves
    if (w == 0 && j + 32 < SEQ) {
      tdm_load_2d(lds_byte_off(&Kb[cur ^ 1][0]),
                  kTile + (size_t)(j + 32) * HDIM, 32 * HDIM, 1, 32 * HDIM);
      tdm_load_2d(lds_byte_off(&Vb[cur ^ 1][0]),
                  vTile + (j + 32), 32, HDIM, SEQ);
    }
    v16bf bk[4];
    bk[0] = load_bfrag(&Kb[cur][(0  + l15) * HDIM + half * 16]);       // kt0 d0-31
    bk[1] = load_bfrag(&Kb[cur][(0  + l15) * HDIM + 32 + half * 16]);  // kt0 d32-63
    bk[2] = load_bfrag(&Kb[cur][(16 + l15) * HDIM + half * 16]);       // kt1 d0-31
    bk[3] = load_bfrag(&Kb[cur][(16 + l15) * HDIM + 32 + half * 16]);  // kt1 d32-63
#else
    const __bf16* kp0 = kb + ((size_t)bh * SEQ + j + l15) * HDIM;
    const __bf16* kp1 = kp0 + 16 * HDIM;
    v16bf bk[4];
    bk[0] = load_bfrag(kp0 + half * 16);
    bk[1] = load_bfrag(kp0 + 32 + half * 16);
    bk[2] = load_bfrag(kp1 + half * 16);
    bk[3] = load_bfrag(kp1 + 32 + half * 16);
#endif
    // ---- logits: two 16x16 tiles covering keys [j, j+32)
    v8f c0 = zero8(), c1 = zero8();
    c0 = bwmma(aq0, bk[0], c0);
    c0 = bwmma(aq1, bk[1], c0);
    c1 = bwmma(aq0, bk[2], c1);
    c1 = bwmma(aq1, bk[3], c1);

    // ---- mask + online softmax (row = qBase + half*8 + r, col = lane&15)
    float p0[8], p1[8], sc[8];
#pragma unroll
    for (int r = 0; r < 8; ++r) {
      const int qrow = qBase + half * 8 + r;
      const unsigned char* mq = mrow + (size_t)qrow * SEQ + j + l15;
      const float v0 = mq[0]  ? -1e10f : c0[r];
      const float v1 = mq[16] ? -1e10f : c1[r];
      const float rm = rowred_max(fmaxf(v0, v1));
      const float mn = fmaxf(mi[r], rm);
      const float e0 = __expf(v0 - mn);
      const float e1 = __expf(v1 - mn);
      const float rs = rowred_sum(e0 + e1);
      const float scl = __expf(mi[r] - mn);
      li[r] = li[r] * scl + rs;
      mi[r] = mn;
      sc[r] = scl;
      p0[r] = e0;
      p1[r] = e1;
    }
#pragma unroll
    for (int nt = 0; nt < 4; ++nt)
#pragma unroll
      for (int r = 0; r < 8; ++r) o[nt][r] *= sc[r];

    // ---- repack P (C-layout) -> A-layout via wave-private LDS (DS in-order)
#pragma unroll
    for (int r = 0; r < 8; ++r) {
      const int row = half * 8 + r;
      pw[row * 32 + l15]      = (__bf16)p0[r];
      pw[row * 32 + 16 + l15] = (__bf16)p1[r];
    }
    const v16bf ap = load_afrag(pw + l15 * 32, half * 16);

    // ---- PV: 4 WMMA (d groups of 16)
    v16bf bv[4];
#if USE_TDM
#pragma unroll
    for (int nt = 0; nt < 4; ++nt)
      bv[nt] = load_bfrag(&Vb[cur][(nt * 16 + l15) * 32 + half * 16]);
#else
    const __bf16* vp = vtb + (size_t)bh * HDIM * SEQ + j + half * 16;
#pragma unroll
    for (int nt = 0; nt < 4; ++nt)
      bv[nt] = load_bfrag(vp + (size_t)(nt * 16 + l15) * SEQ);
#endif
#pragma unroll
    for (int nt = 0; nt < 4; ++nt)
      o[nt] = bwmma(ap, bv[nt], o[nt]);
#if USE_TDM
    __syncthreads();                       // done reading buf[cur]
#endif
  }

  // ---- epilogue: divide by row sums, scatter ctx[t, h*64+d] as bf16
  float invl[8];
#pragma unroll
  for (int r = 0; r < 8; ++r) invl[r] = 1.f / li[r];
#pragma unroll
  for (int nt = 0; nt < 4; ++nt) {
    const int f = h * HDIM + nt * 16 + l15;
#pragma unroll
    for (int r = 0; r < 8; ++r) {
      const int qrow = qBase + half * 8 + r;
      const int tkn  = qrow * BATCH + b;
      ctxb[(size_t)tkn * HIDDEN + f] = (__bf16)(o[nt][r] * invl[r]);
    }
  }
}

// ---------------------------------------------------------------------------
// Kernel 5: out-projection GEMM. out[4096 x 1024] = ctx * WoutT^T (fp32 out).
// ---------------------------------------------------------------------------
__global__ __launch_bounds__(256)
void out_gemm_kernel(const __bf16* __restrict__ A, const __bf16* __restrict__ Bt,
                     float* __restrict__ out) {
  const int lane = threadIdx.x & 31, w = threadIdx.x >> 5;
  const int half = lane >> 4, l15 = lane & 15;
  const int mBlock = blockIdx.y * 32;
  const int nBlock = blockIdx.x * 512;

  v8f acc[8] = {zero8(), zero8(), zero8(), zero8(),
                zero8(), zero8(), zero8(), zero8()};
  gemm_core(A, Bt, mBlock, nBlock, acc);

  const int mBase = mBlock + (w >> 2) * 16;
  const int nBase = nBlock + (w & 3) * 128;
#pragma unroll
  for (int nt = 0; nt < 8; ++nt) {
    const int n = nBase + nt * 16 + l15;
#pragma unroll
    for (int r = 0; r < 8; ++r) {
      const int m = mBase + half * 8 + r;
      out[(size_t)m * HIDDEN + n] = acc[nt][r];
    }
  }
}

// ---------------------------------------------------------------------------
extern "C" void kernel_launch(void* const* d_in, const int* in_sizes, int n_in,
                              void* d_out, int out_size, void* d_ws, size_t ws_size,
                              hipStream_t stream) {
  (void)in_sizes; (void)n_in; (void)out_size; (void)ws_size;
  const float*         inputs_q = (const float*)d_in[0];
  const unsigned char* mask     = (const unsigned char*)d_in[1];
  const float*         ln_scale = (const float*)d_in[2];
  const float*         ln_bias  = (const float*)d_in[3];
  const float*         w_qkv    = (const float*)d_in[4];
  const float*         w_out    = (const float*)d_in[5];
  float*               out      = (float*)d_out;

  // Workspace layout (bf16), total ~48 MB.
  __bf16* lnb   = (__bf16*)d_ws;                         // 4096*1024
  __bf16* wqkvT = lnb   + (size_t)NTOK * HIDDEN;         // 3072*1024
  __bf16* woutT = wqkvT + (size_t)QKVN * HIDDEN;         // 1024*1024
  __bf16* qb    = woutT + (size_t)HIDDEN * HIDDEN;       // 2*16*2048*64
  __bf16* kb    = qb    + (size_t)NTOK * HIDDEN;
  __bf16* vtb   = kb    + (size_t)NTOK * HIDDEN;
  __bf16* ctxb  = vtb   + (size_t)NTOK * HIDDEN;

  ln_bf16_kernel<<<NTOK, 256, 0, stream>>>(inputs_q, ln_scale, ln_bias, lnb);
  wqkv_T_kernel<<<(QKVN * HIDDEN + 255) / 256, 256, 0, stream>>>(w_qkv, wqkvT);
  wout_T_kernel<<<(HIDDEN * HIDDEN + 255) / 256, 256, 0, stream>>>(w_out, woutT);
  qkv_gemm_kernel<<<dim3(QKVN / 512, NTOK / 32), 256, 0, stream>>>(lnb, wqkvT, qb, kb, vtb);
  attn_kernel<<<(BATCH * NHEAD * (SEQ / 16)) / 8, 256, 0, stream>>>(qb, kb, vtb, mask, ctxb);
  out_gemm_kernel<<<dim3(HIDDEN / 512, NTOK / 32), 256, 0, stream>>>(ctxb, woutT, out);
}
